// CommunicationNet_43911745634318
// MI455X (gfx1250) — matrix operand order repaired and checked by reading, not compiled, not run on hardware
//
#include <hip/hip_runtime.h>

typedef __attribute__((ext_vector_type(16))) _Float16 v16h;
typedef __attribute__((ext_vector_type(8)))  _Float16 v8h;
typedef __attribute__((ext_vector_type(8)))  float    v8f;

#define N_RUNS  128
#define T_STEPS 2048
#define NA      16
#define SENS    7
#define HID     22
#define RPB     16   // runs per block == WMMA M

#if defined(__gfx1250__) && __has_builtin(__builtin_amdgcn_global_load_async_to_lds_b128)
#define HAVE_ASYNC_LDS 1
#else
#define HAVE_ASYNC_LDS 0
#endif

__device__ __forceinline__ void wait_async_le14() {
#if defined(__gfx1250__)
#if __has_builtin(__builtin_amdgcn_s_wait_asynccnt)
  __builtin_amdgcn_s_wait_asynccnt(14);
#else
  asm volatile("s_wait_asynccnt 0xe" ::: "memory");
#endif
#endif
}
__device__ __forceinline__ void wait_async_0() {
#if defined(__gfx1250__)
#if __has_builtin(__builtin_amdgcn_s_wait_asynccnt)
  __builtin_amdgcn_s_wait_asynccnt(0);
#else
  asm volatile("s_wait_asynccnt 0x0" ::: "memory");
#endif
#endif
}

// Stage 224 bytes (this lane's half-run of one timestep): 14 x b128 global->LDS.
__device__ __forceinline__ void stage_sens(const float* gsrc, float* ldst) {
#if HAVE_ASYNC_LDS
  typedef int v4i_vs __attribute__((vector_size(16)));
  v4i_vs* g = (v4i_vs*)gsrc;  // C-style cast drops const; generic AS pointer
  __attribute__((address_space(3))) v4i_vs* l =
      (__attribute__((address_space(3))) v4i_vs*)ldst;
  #pragma unroll
  for (int q = 0; q < 14; ++q)
    __builtin_amdgcn_global_load_async_to_lds_b128(g + q, l + q, 0, 0);
#else
  const float4* src = (const float4*)gsrc;
  float4* dst = (float4*)ldst;
  #pragma unroll
  for (int q = 0; q < 14; ++q)
    dst[q] = src[q];
#endif
}

__global__ __launch_bounds__(32)
void comm_net_wmma_kernel(const float* __restrict__ runs,
                          const float* __restrict__ comm_init,
                          const float* __restrict__ W1,
                          const float* __restrict__ b1,
                          const float* __restrict__ W2,
                          const float* __restrict__ b2,
                          float* __restrict__ out)
{
  __shared__ __align__(16) float    sens[2][RPB][NA * SENS]; // double-buffered sensing
  __shared__ float                  commL[RPB][NA + 2];      // recurrent comm state (f32)
  __shared__ __align__(16) _Float16 hbuf[RPB][24];           // hidden acts, padded 22->24
  __shared__ __align__(16) float    ctrl[RPB][NA];           // controls for one timestep

  const int lane = threadIdx.x;
  const int half = lane >> 4;   // 0: lanes 0-15, 1: lanes 16-31
  const int lm   = lane & 15;
  const int r0   = blockIdx.x * RPB;

  // ---- constant weight fragments (B operand: B[k][n]; lane holds col n,
  //      element j holds k = (j<8 ? j : j+8) + half*8) ----
  v16h B1a, B1b, B2f;
  #pragma unroll
  for (int j = 0; j < 16; ++j) {
    int k  = (j < 8 ? j : j + 8) + half * 8;
    int n1 = 16 + lm;
    B1a[j] = (_Float16)((k < SENS + 2) ? W1[lm * (SENS + 2) + k] : 0.0f);
    B1b[j] = (_Float16)((k < SENS + 2 && n1 < HID) ? W1[n1 * (SENS + 2) + k] : 0.0f);
    B2f[j] = (_Float16)((lm < 2 && k < HID) ? W2[lm * HID + k] : 0.0f);
  }
  const float b1t0 = b1[lm];
  const float b1t1 = (16 + lm < HID) ? b1[16 + lm] : 0.0f;
  const float b2n  = (lm < 2) ? b2[lm] : 0.0f;

  // ---- init comm state + hbuf zero padding ----
  for (int idx = lane; idx < RPB * (NA + 2); idx += 32)
    commL[idx / (NA + 2)][idx % (NA + 2)] =
        comm_init[(size_t)(r0 + idx / (NA + 2)) * (NA + 2) + idx % (NA + 2)];
  if (lane < RPB) { hbuf[lane][22] = (_Float16)0.0f; hbuf[lane][23] = (_Float16)0.0f; }

  const int rr = lane >> 1;   // staging: 2 lanes per run (224 B each)
  const int hh = lane & 1;
  const float* gbase = runs + ((size_t)(r0 + rr) * T_STEPS) * (NA * SENS) + hh * 56;

  // prologue: stage t=0, drain, publish
  stage_sens(gbase, &sens[0][rr][hh * 56]);
  wait_async_0();
  __syncthreads();

  for (int t = 0; t < T_STEPS; ++t) {
    const int cur = t & 1;

    // ---- async-stage next timestep into the other buffer, keep it in flight ----
    if (t + 1 < T_STEPS) {
      stage_sens(gbase + (size_t)(t + 1) * (NA * SENS), &sens[cur ^ 1][rr][hh * 56]);
      if (t + 2 < T_STEPS) {   // warm L2 two steps ahead
        const char* np = (const char*)(gbase + (size_t)(t + 2) * (NA * SENS));
        __builtin_prefetch(np, 0, 0);
        __builtin_prefetch(np + 128, 0, 0);
      }
      wait_async_le14();       // all but the newest 14 (t+1's) complete => buf[cur] ready
    } else {
      wait_async_0();
    }
    __syncthreads();

    // ---- sequential agent recurrence (reference Sync.sequential semantics) ----
    for (int i = 0; i < NA; ++i) {
      // A operand: lane = row (run), element j -> k = (j<8 ? j : j+8) + half*8
      v16h A1 = {};
      if (half == 0) {
        const float* sp = &sens[cur][lm][i * SENS];
        #pragma unroll
        for (int s = 0; s < SENS; ++s) A1[s] = (_Float16)sp[s];  // k = 0..6
        A1[7] = (_Float16)commL[lm][i];                          // k = 7
      } else {
        A1[0] = (_Float16)commL[lm][i + 2];                      // k = 8
      }

      v8f d0 = {}, d1 = {};
      d0 = __builtin_amdgcn_wmma_f32_16x16x32_f16(false, A1, false, B1a, (short)0, d0, false, false);
      d1 = __builtin_amdgcn_wmma_f32_16x16x32_f16(false, A1, false, B1b, (short)0, d1, false, false);

      // relu(d + b1) -> hbuf[m][n] in f16; lane owns col n, rows half*8 + m
      #pragma unroll
      for (int m = 0; m < 8; ++m)
        hbuf[half * 8 + m][lm] = (_Float16)fmaxf(d0[m] + b1t0, 0.0f);
      if (lm < HID - 16) {
        #pragma unroll
        for (int m = 0; m < 8; ++m)
          hbuf[half * 8 + m][16 + lm] = (_Float16)fmaxf(d1[m] + b1t1, 0.0f);
      }
      __syncthreads();

      // A2: row lm, same k mapping -> two aligned 8-half LDS chunks
      v16h A2 = {};
      {
        v8h lo = *(const v8h*)&hbuf[lm][half * 8];
        #pragma unroll
        for (int j = 0; j < 8; ++j) A2[j] = lo[j];
        if (half == 0) {                       // k = 16..23 (22,23 zero-padded)
          v8h hi = *(const v8h*)&hbuf[lm][16];
          #pragma unroll
          for (int j = 0; j < 8; ++j) A2[8 + j] = hi[j];
        }                                      // half==1: k = 24..31 -> zeros
      }

      v8f d2 = {};
      d2 = __builtin_amdgcn_wmma_f32_16x16x32_f16(false, A2, false, B2f, (short)0, d2, false, false);

      // col 0 = control, col 1 = new comm; lane n==lm owns rows half*8 + m
      if (lm == 1) {
        #pragma unroll
        for (int m = 0; m < 8; ++m)
          commL[half * 8 + m][i + 1] = d2[m] + b2n;   // written before agent i+1 reads it
      } else if (lm == 0) {
        #pragma unroll
        for (int m = 0; m < 8; ++m)
          ctrl[half * 8 + m][i] = d2[m] + b2n;
      }
      __syncthreads();
    }

    // ---- coalesced store of this timestep's controls ----
    for (int idx = lane; idx < RPB * NA / 4; idx += 32) {
      int r2 = idx >> 2, q = idx & 3;
      float4 v = ((const float4*)&ctrl[r2][0])[q];
      *(float4*)&out[((size_t)(r0 + r2) * T_STEPS + t) * NA + q * 4] = v;
    }
    __syncthreads();
  }
}

extern "C" void kernel_launch(void* const* d_in, const int* in_sizes, int n_in,
                              void* d_out, int out_size, void* d_ws, size_t ws_size,
                              hipStream_t stream) {
  (void)in_sizes; (void)n_in; (void)d_ws; (void)ws_size; (void)out_size;
  const float* runs      = (const float*)d_in[0];
  const float* comm_init = (const float*)d_in[1];
  const float* W1        = (const float*)d_in[2];
  const float* b1        = (const float*)d_in[3];
  const float* W2        = (const float*)d_in[4];
  const float* b2        = (const float*)d_in[5];
  float* out = (float*)d_out;

  dim3 grid(N_RUNS / RPB);   // 8 blocks, one wave32 each, 16 runs per block
  dim3 block(32);
  hipLaunchKernelGGL(comm_net_wmma_kernel, grid, block, 0, stream,
                     runs, comm_init, W1, b1, W2, b2, out);
}
